// BiAttention_16381005267228
// MI455X (gfx1250) — matrix operand order repaired and checked by reading, compiled, not run on hardware
//
#include <hip/hip_runtime.h>
#include <cstdint>

#define B_   16
#define QL_  512
#define CL_  2048
#define D_   2048
#define D3_  (3 * D_)
#define APITCH 520   // alpha LDS row pitch (ushorts): 1040B rows -> 2-way-optimal b128 banks
#define TT 64        // transpose tile dim
#define TP 80        // transpose LDS pitch (ushorts), keeps us16 stores 32B-aligned

typedef __attribute__((ext_vector_type(8)))  unsigned short us8;
typedef __attribute__((ext_vector_type(16))) unsigned short us16;
typedef __attribute__((ext_vector_type(16))) __bf16         v16bf;
typedef __attribute__((ext_vector_type(8)))  float          v8f;

static __device__ __forceinline__ unsigned short f2bf(float f) {
    unsigned int u = __float_as_uint(f);
    u += 0x7FFFu + ((u >> 16) & 1u);           // round-to-nearest-even
    return (unsigned short)(u >> 16);
}

static __device__ __forceinline__ v8f wmma_bf16(us16 a, us16 b, v8f c) {
    return __builtin_amdgcn_wmma_f32_16x16x32_bf16(
        false, __builtin_bit_cast(v16bf, a),
        false, __builtin_bit_cast(v16bf, b),
        (short)0, c, false, false);
}

// ---------------------------------------------------------------------------
// prep_q: one wave per (b,j) row. Writes q_bf (row-major) and sq = question.wq
// ---------------------------------------------------------------------------
__global__ __launch_bounds__(256) void prep_q(const float* __restrict__ q,
                                              const float* __restrict__ wsim,
                                              unsigned short* __restrict__ q_bf,
                                              float* __restrict__ sq) {
    const int gw   = (blockIdx.x * 256 + threadIdx.x) >> 5;   // row index, < B*QL
    const int lane = threadIdx.x & 31;
    const float* row = q + (size_t)gw * D_;
    unsigned short* orow = q_bf + (size_t)gw * D_;
    const float* wq = wsim;
    float acc = 0.f;
    for (int c = 0; c < D_ / 32; ++c) {
        const int d = c * 32 + lane;
        const float v = row[d];
        acc += v * wq[d];
        orow[d] = f2bf(v);
    }
#pragma unroll
    for (int o = 16; o > 0; o >>= 1) acc += __shfl_xor(acc, o, 32);
    if (lane == 0) sq[gw] = acc;
}

// ---------------------------------------------------------------------------
// transpose_q: LDS-tiled 64x64 bf16 transpose q_bf[j][d] -> qT_bf[d][j].
// Coalesced 32B global loads/stores; shuffle through LDS.
// ---------------------------------------------------------------------------
__global__ __launch_bounds__(256) void transpose_q(const unsigned short* __restrict__ q_bf,
                                                   unsigned short* __restrict__ qT_bf) {
    __shared__ unsigned short tile[TT * TP];
    const int d0 = blockIdx.x * TT;          // gridDim.x = D/64
    const int j0 = blockIdx.y * TT;          // gridDim.y = QL/64
    const int b  = blockIdx.z;               // gridDim.z = B
    const int rr = threadIdx.x >> 2;         // 0..63
    const int cc = (threadIdx.x & 3) << 4;   // 0,16,32,48

    const unsigned short* src = q_bf + ((size_t)b * QL_ + j0 + rr) * D_ + d0 + cc;
    *(us16*)&tile[rr * TP + cc] = *(const us16*)src;
    __syncthreads();

    us16 vv;
#pragma unroll
    for (int k = 0; k < 16; ++k) vv[k] = tile[(cc + k) * TP + rr];
    unsigned short* dst = qT_bf + ((size_t)b * D_ + d0 + rr) * QL_ + j0 + cc;
    *(us16*)dst = vv;
}

// ---------------------------------------------------------------------------
// prep_c: one wave per (b,i) row. Writes cm_bf = bf16(context*wm),
// sc = context.wc, and out[...,0:D] = context
// ---------------------------------------------------------------------------
__global__ __launch_bounds__(256) void prep_c(const float* __restrict__ ctx,
                                              const float* __restrict__ wsim,
                                              unsigned short* __restrict__ cm_bf,
                                              float* __restrict__ sc,
                                              float* __restrict__ out) {
    const int gw   = (blockIdx.x * 256 + threadIdx.x) >> 5;   // row index, < B*CL
    const int lane = threadIdx.x & 31;
    const float* row = ctx + (size_t)gw * D_;
    const float* wc = wsim + D_;
    const float* wm = wsim + 2 * D_;
    float acc = 0.f;
    for (int c = 0; c < D_ / 32; ++c) {
        const int d = c * 32 + lane;
        const float v = row[d];
        acc += v * wc[d];
        cm_bf[(size_t)gw * D_ + d] = f2bf(v * wm[d]);
        out[(size_t)gw * D3_ + d]  = v;                       // part 1: context copy
    }
#pragma unroll
    for (int o = 16; o > 0; o >>= 1) acc += __shfl_xor(acc, o, 32);
    if (lane == 0) sc[gw] = acc;
}

// ---------------------------------------------------------------------------
// attn: one wave per WG, a 32-row strip of context (M=32 register blocking:
// two A-frags share every B-frag -> 2 WMMAs per 1KB B fetch, 2x arithmetic
// intensity vs M=16).
//   GEMM1: S[32,512] = cm_bf @ q_bfT            (512 f32 acc VGPRs)
//   softmax over j (sq folded; sc cancels in softmax_j); rowM = max_j S + sc
//   GEMM2: c2q[32,2048] = alpha(bf16, LDS) @ question (qT_bf), fused epilogue
//          out[...,D:2D] = context * c2q
// ---------------------------------------------------------------------------
__global__ __launch_bounds__(32) void attn_kernel(
    const float* __restrict__ ctx,
    const unsigned short* __restrict__ qbf,
    const unsigned short* __restrict__ qtbf,
    const unsigned short* __restrict__ cmbf,
    const float* __restrict__ sqg,
    const float* __restrict__ scg,
    float* __restrict__ rowMg,
    float* __restrict__ out) {
    __shared__ unsigned short aLDS[32 * APITCH];   // 33,280 B

    const int lane = threadIdx.x & 31;
    const int ln   = lane & 15;     // A: row m / B: col n / C/D: col n
    const int kh   = lane >> 4;     // K-half select

    const int rb = blockIdx.x;                // strip id, < B*CL/32
    const int b  = rb >> 6;
    const int i0 = (rb & 63) << 5;

    const unsigned short* cmb = cmbf + ((size_t)b * CL_ + i0) * D_;
    const unsigned short* qb  = qbf  + (size_t)b * QL_ * D_;
    const unsigned short* qtb = qtbf + (size_t)b * D_ * QL_;
    const float* sqp   = sqg + b * QL_;
    const float* scp   = scg + (size_t)b * CL_ + i0;
    float*       rowMp = rowMg + (size_t)b * CL_ + i0;

    v8f acc[2][32];
#pragma unroll
    for (int g = 0; g < 2; ++g)
#pragma unroll
        for (int t = 0; t < 32; ++t) acc[g][t] = (v8f){0, 0, 0, 0, 0, 0, 0, 0};

    // per-lane fragment bases (ISA 16-bit A/B VGPR layouts -> contiguous chunks)
    const unsigned short* pa0 = cmb + (size_t)ln * D_ + 8 * kh;          // rows 0..15
    const unsigned short* pa1 = cmb + (size_t)(16 + ln) * D_ + 8 * kh;   // rows 16..31
    const unsigned short* pb0 = qb  + (size_t)ln * D_ + 8 * kh;

#pragma unroll 1
    for (int kc = 0; kc < D_ / 32; ++kc) {
        __builtin_prefetch(pa0 + (kc + 8) * 32, 0, 0);
        __builtin_prefetch(pa1 + (kc + 8) * 32, 0, 0);
        const us8 a0lo = *(const us8*)(pa0 + kc * 32);
        const us8 a0hi = *(const us8*)(pa0 + kc * 32 + 16);
        const us8 a1lo = *(const us8*)(pa1 + kc * 32);
        const us8 a1hi = *(const us8*)(pa1 + kc * 32 + 16);
        const us16 a0 = __builtin_shufflevector(a0lo, a0hi,
            0, 1, 2, 3, 4, 5, 6, 7, 8, 9, 10, 11, 12, 13, 14, 15);
        const us16 a1 = __builtin_shufflevector(a1lo, a1hi,
            0, 1, 2, 3, 4, 5, 6, 7, 8, 9, 10, 11, 12, 13, 14, 15);
#pragma unroll
        for (int t = 0; t < 32; ++t) {
            const us8 blo = *(const us8*)(pb0 + (size_t)t * 16 * D_ + kc * 32);
            const us8 bhi = *(const us8*)(pb0 + (size_t)t * 16 * D_ + kc * 32 + 16);
            const us16 bf = __builtin_shufflevector(blo, bhi,
                0, 1, 2, 3, 4, 5, 6, 7, 8, 9, 10, 11, 12, 13, 14, 15);
            acc[0][t] = wmma_bf16(a0, bf, acc[0][t]);
            acc[1][t] = wmma_bf16(a1, bf, acc[1][t]);
        }
    }

    // fold sq[j] into S (sc[i] cancels within softmax over j)
#pragma unroll
    for (int t = 0; t < 32; ++t) {
        const float sv = sqp[16 * t + ln];
#pragma unroll
        for (int r = 0; r < 8; ++r) {
            acc[0][t][r] += sv;
            acc[1][t][r] += sv;
        }
    }

#pragma unroll
    for (int g = 0; g < 2; ++g) {
#pragma unroll
        for (int r = 0; r < 8; ++r) {
            float mx = -3.402823466e38f;
#pragma unroll
            for (int t = 0; t < 32; ++t) mx = fmaxf(mx, acc[g][t][r]);
            mx = fmaxf(mx, __shfl_xor(mx, 1, 32));
            mx = fmaxf(mx, __shfl_xor(mx, 2, 32));
            mx = fmaxf(mx, __shfl_xor(mx, 4, 32));
            mx = fmaxf(mx, __shfl_xor(mx, 8, 32));
            float s = 0.f;
#pragma unroll
            for (int t = 0; t < 32; ++t) {
                const float e = __expf(acc[g][t][r] - mx);
                acc[g][t][r] = e;
                s += e;
            }
            s += __shfl_xor(s, 1, 32);
            s += __shfl_xor(s, 2, 32);
            s += __shfl_xor(s, 4, 32);
            s += __shfl_xor(s, 8, 32);
            const float inv = 1.0f / s;
            const int row = 16 * g + r + 8 * kh;      // C/D layout: M = r + 8*khalf
            if (ln == 0) rowMp[row] = mx + scp[row];  // full S row-max incl. sc
#pragma unroll
            for (int t = 0; t < 32; ++t)
                aLDS[row * APITCH + 16 * t + ln] = f2bf(acc[g][t][r] * inv);
        }
    }

    __syncthreads();   // single-wave WG: lowers to S_NOP, LDS stays in-order

    // GEMM2: c2q = alpha @ question; A from LDS, B from qT_bf (contiguous)
    const unsigned short* ap0 = aLDS + (size_t)ln * APITCH + 8 * kh;
    const unsigned short* ap1 = aLDS + (size_t)(16 + ln) * APITCH + 8 * kh;
#pragma unroll 1
    for (int nt = 0; nt < D_ / 16; ++nt) {
        v8f o0 = (v8f){0, 0, 0, 0, 0, 0, 0, 0};
        v8f o1 = (v8f){0, 0, 0, 0, 0, 0, 0, 0};
        const unsigned short* pbt = qtb + (size_t)(16 * nt + ln) * QL_ + 8 * kh;
#pragma unroll
        for (int kc = 0; kc < QL_ / 32; ++kc) {
            const us8 a0lo = *(const us8*)(ap0 + kc * 32);
            const us8 a0hi = *(const us8*)(ap0 + kc * 32 + 16);
            const us8 a1lo = *(const us8*)(ap1 + kc * 32);
            const us8 a1hi = *(const us8*)(ap1 + kc * 32 + 16);
            const us16 a0 = __builtin_shufflevector(a0lo, a0hi,
                0, 1, 2, 3, 4, 5, 6, 7, 8, 9, 10, 11, 12, 13, 14, 15);
            const us16 a1 = __builtin_shufflevector(a1lo, a1hi,
                0, 1, 2, 3, 4, 5, 6, 7, 8, 9, 10, 11, 12, 13, 14, 15);
            const us8 blo = *(const us8*)(pbt + kc * 32);
            const us8 bhi = *(const us8*)(pbt + kc * 32 + 16);
            const us16 bf = __builtin_shufflevector(blo, bhi,
                0, 1, 2, 3, 4, 5, 6, 7, 8, 9, 10, 11, 12, 13, 14, 15);
            o0 = wmma_bf16(a0, bf, o0);
            o1 = wmma_bf16(a1, bf, o1);
        }
        const int col = nt * 16 + ln;
#pragma unroll
        for (int r = 0; r < 8; ++r) {
            const size_t row0 = (size_t)b * CL_ + i0 + r + 8 * kh;
            const size_t row1 = row0 + 16;
            out[row0 * D3_ + D_ + col] = ctx[row0 * D_ + col] * o0[r];   // part 2
            out[row1 * D3_ + D_ + col] = ctx[row1 * D_ + col] * o1[r];
        }
    }
}

// ---------------------------------------------------------------------------
// beta: per batch, softmax over i of rowM (= sc + max_j S)
// ---------------------------------------------------------------------------
__global__ __launch_bounds__(256) void beta_kernel(const float* __restrict__ rowM,
                                                   float* __restrict__ beta) {
    __shared__ float sh[256];
    const int b = blockIdx.x, tid = threadIdx.x;
    const float* rm = rowM + (size_t)b * CL_;
    float* bt = beta + (size_t)b * CL_;
    float lm = -3.402823466e38f;
    for (int i = tid; i < CL_; i += 256) lm = fmaxf(lm, rm[i]);
    sh[tid] = lm;
    __syncthreads();
    for (int o = 128; o > 0; o >>= 1) {
        if (tid < o) sh[tid] = fmaxf(sh[tid], sh[tid + o]);
        __syncthreads();
    }
    const float mx = sh[0];
    __syncthreads();
    float ls = 0.f;
    for (int i = tid; i < CL_; i += 256) {
        const float e = __expf(rm[i] - mx);
        bt[i] = e;
        ls += e;
    }
    sh[tid] = ls;
    __syncthreads();
    for (int o = 128; o > 0; o >>= 1) {
        if (tid < o) sh[tid] += sh[tid + o];
        __syncthreads();
    }
    const float inv = 1.0f / sh[0];
    for (int i = tid; i < CL_; i += 256) bt[i] *= inv;
}

// ---------------------------------------------------------------------------
// q2c: q2c[b,d] = sum_i beta[b,i] * context[b,i,d]
// ---------------------------------------------------------------------------
__global__ __launch_bounds__(256) void q2c_kernel(const float* __restrict__ ctx,
                                                  const float* __restrict__ beta,
                                                  float* __restrict__ q2c) {
    const int b = blockIdx.x >> 3;
    const int d = ((blockIdx.x & 7) << 8) + threadIdx.x;
    const float* cb = ctx + (size_t)b * CL_ * D_ + d;
    const float* bt = beta + (size_t)b * CL_;
    float acc = 0.f;
    for (int i = 0; i < CL_; ++i) acc += bt[i] * cb[(size_t)i * D_];
    q2c[(size_t)b * D_ + d] = acc;
}

// ---------------------------------------------------------------------------
// part3: out[...,2D:3D] = context * q2c (broadcast over i)
// ---------------------------------------------------------------------------
__global__ __launch_bounds__(256) void part3_kernel(const float* __restrict__ ctx,
                                                    const float* __restrict__ q2c,
                                                    float* __restrict__ out) {
    const size_t total = (size_t)B_ * CL_ * D_;
    for (size_t idx = (size_t)blockIdx.x * 256 + threadIdx.x; idx < total;
         idx += (size_t)gridDim.x * 256) {
        const size_t b   = idx / ((size_t)CL_ * D_);
        const size_t rem = idx - b * ((size_t)CL_ * D_);
        const size_t i   = rem / D_;
        const size_t d   = rem - i * D_;
        out[(b * CL_ + i) * D3_ + 2 * D_ + d] = ctx[idx] * q2c[b * D_ + d];
    }
}

// ---------------------------------------------------------------------------
extern "C" void kernel_launch(void* const* d_in, const int* in_sizes, int n_in,
                              void* d_out, int out_size, void* d_ws, size_t ws_size,
                              hipStream_t stream) {
    (void)in_sizes; (void)n_in; (void)out_size; (void)ws_size;
    const float* q    = (const float*)d_in[0];
    const float* ctx  = (const float*)d_in[1];
    const float* wsim = (const float*)d_in[2];
    float* out = (float*)d_out;

    char* ws = (char*)d_ws;   // ~192.5 MB used
    unsigned short* q_bf  = (unsigned short*)(ws);                 //  32 MB
    unsigned short* qT_bf = (unsigned short*)(ws + 33554432ull);   //  32 MB
    unsigned short* cm_bf = (unsigned short*)(ws + 67108864ull);   // 128 MB
    float* sq   = (float*)(ws + 201326592ull);                     //  32 KB
    float* sc   = (float*)(ws + 201359360ull);                     // 128 KB
    float* rowM = (float*)(ws + 201490432ull);                     // 128 KB
    float* beta = (float*)(ws + 201621504ull);                     // 128 KB
    float* q2c  = (float*)(ws + 201752576ull);                     // 128 KB

    prep_q<<<(B_ * QL_) / 8, 256, 0, stream>>>(q, wsim, q_bf, sq);
    transpose_q<<<dim3(D_ / TT, QL_ / TT, B_), 256, 0, stream>>>(q_bf, qT_bf);
    prep_c<<<(B_ * CL_) / 8, 256, 0, stream>>>(ctx, wsim, cm_bf, sc, out);
    attn_kernel<<<B_ * CL_ / 32, 32, 0, stream>>>(ctx, q_bf, qT_bf, cm_bf,
                                                  sq, sc, rowM, out);
    beta_kernel<<<B_, 256, 0, stream>>>(rowM, beta);
    q2c_kernel<<<B_ * (D_ / 256), 256, 0, stream>>>(ctx, beta, q2c);
    part3_kernel<<<4096, 256, 0, stream>>>(ctx, q2c, out);
}
